// MLPExperts_28767690949044
// MI455X (gfx1250) — matrix operand order repaired and checked by reading, not compiled, not run on hardware
//
#include <hip/hip_runtime.h>

// MoE expert MLP, fused:  out = relu(LN(x@W1+b1)) @ W2 + b2   (L=1)
// E=16, N=16384, D=O=512.  bf16 WMMA (16x16x32), fp32 accumulation.
//
// Pass 1 (tiny): convert+swizzle fp32 weights -> bf16 WMMA-B-fragment layout in d_ws
//                (needs 16 MiB of workspace; weights then live in L2).
// Pass 2 (main): per 32-token block: GEMM1 -> bias -> LN(from accumulators) -> relu
//                -> bf16 A-panel in LDS -> GEMM2 -> bias -> fp32 out.
//                Weight slabs staged LDS-side via global_load_async_to_lds_b128
//                (ASYNCcnt), fragment reads are 2x ds_load_b128, conflict-free.

#define EXPERTS 16
#define NTOK    16384
#define DDIM    512
#define ODIM    512
#define ROWS    32          // token rows per block
#define KB      32          // K-slab per WMMA step (bf16 K=32)
#define NSLAB   (DDIM / KB) // 16 slabs
#define LREC    40          // padded shorts per lane record (data=16) -> 80B, 16B aligned
#define SLAB_BYTES (32 * 32 * 16 * 2)   // 32 tiles * 32 lanes * 16 bf16 = 32 KiB (unpadded, global)

typedef __attribute__((ext_vector_type(16))) __bf16 v16bf;
typedef __attribute__((ext_vector_type(8)))  float  v8f;

union AF { v16bf v; unsigned int u[8]; };

__device__ __forceinline__ unsigned short f2bf(float f) {
    unsigned int x = __float_as_uint(f);
    x += 0x7FFFu + ((x >> 16) & 1u);          // round-to-nearest-even
    return (unsigned short)(x >> 16);
}
__device__ __forceinline__ unsigned int pack2(float lo, float hi) {
    return (unsigned int)f2bf(lo) | ((unsigned int)f2bf(hi) << 16);
}

// ---------------- pass 1: weight convert + B-fragment swizzle ----------------
// ws layout (shorts): region r = which*E*16 + e*16 + ks   (which: 0=W1, 1=W2)
//   ((r*32 + tile)*32 + lane)*16 + s ;  lane = (f&15) + 16*(klocal>=16), s = klocal&15
__global__ __launch_bounds__(256)
void swizzle_weights(const float* __restrict__ Wm, const float* __restrict__ Wo,
                     unsigned short* __restrict__ ws)
{
    size_t i = (size_t)blockIdx.x * 256 + threadIdx.x;       // source element id
    const size_t PER = (size_t)EXPERTS * DDIM * DDIM;
    if (i >= 2 * PER) return;
    int which = i >= PER;
    size_t j = which ? i - PER : i;
    int f = (int)(j & (DDIM - 1));
    size_t r = j >> 9;
    int k = (int)(r & (DDIM - 1));
    int e = (int)(r >> 9);
    float v = (which ? Wo : Wm)[j];
    int ks = k >> 5, kl = k & 31;
    int t  = f >> 4;
    int l  = (f & 15) + (kl & 16);
    int s  = kl & 15;
    size_t region = (size_t)(which ? EXPERTS * NSLAB : 0) + (size_t)e * NSLAB + ks;
    ws[((region * 32 + t) * 32 + l) * 16 + s] = f2bf(v);
}

// ---------------- pass 2: fused MLP ----------------
__global__ __launch_bounds__(256)
void moe_mlp_fused(const float* __restrict__ x,
                   const unsigned short* __restrict__ wsw,   // swizzled bf16 weights
                   const float* __restrict__ bm,  const float* __restrict__ bo,
                   const float* __restrict__ gamma, const float* __restrict__ beta,
                   float* __restrict__ out)
{
    // A panel: 32 rows x 512 K as bf16 pairs. Row stride 258 uints -> conflict-free gathers.
    __shared__ unsigned int   sA[ROWS][DDIM / 2 + 2];
    // B slab: 32 tiles x 32 lanes x LREC shorts (data 16). 80B lane stride -> conflict-free b128.
    __shared__ unsigned short sBs[32 * 32 * LREC];
    __shared__ float sG[DDIM], sBt[DDIM];
    __shared__ float sPart[ROWS][4][2];
    __shared__ float sMu[ROWS], sRstd[ROWS];

    const int tid  = threadIdx.x;
    const int lane = tid & 31;
    const int wave = tid >> 5;
    const int wm   = wave & 1;        // 16-row M tile
    const int wq   = wave >> 1;       // 128-col quarter
    const int hi   = lane >> 4;
    const int ln15 = lane & 15;
    const int e    = blockIdx.y;
    const int row0 = blockIdx.x * ROWS;

    const unsigned  sbB   = (unsigned)(uintptr_t)&sBs[0];     // LDS byte offset (low 32 bits)
    const unsigned long long wbase = (unsigned long long)(uintptr_t)wsw;

    for (int i = tid; i < DDIM; i += 256) { sG[i] = gamma[i]; sBt[i] = beta[i]; }

    // stage x rows -> bf16 A panel
    const float* xrow = x + ((size_t)e * NTOK + row0) * DDIM;
    for (int p = tid; p < ROWS * (DDIM / 2); p += 256) {
        int r  = p >> 8;
        int kp = p & 255;
        float2 v = *(const float2*)(xrow + (size_t)r * DDIM + 2 * kp);
        sA[r][kp] = pack2(v.x, v.y);
    }

    v8f acc[8];
#pragma unroll
    for (int t = 0; t < 8; ++t)
#pragma unroll
        for (int r = 0; r < 8; ++r) acc[t][r] = 0.f;

    // ---- one GEMM over the current A panel, weights from swizzled region ----
    auto run_gemm = [&](int regionBase) {
        for (int ks = 0; ks < NSLAB; ++ks) {
            __syncthreads();                      // previous slab fully consumed
            {
                unsigned slab = (unsigned)((regionBase + ks) * SLAB_BYTES);
#pragma unroll
                for (int j = 0; j < 8; ++j) {
                    int c  = tid + 256 * j;       // 16B chunk id within slab (0..2047)
                    int t  = c >> 6;
                    int lr = (c >> 1) & 31;
                    int hf = c & 1;
                    unsigned dst = sbB + (unsigned)(t * (32 * LREC * 2) + lr * (LREC * 2) + hf * 16);
                    unsigned src = slab + (unsigned)(c * 16);
                    asm volatile("global_load_async_to_lds_b128 %0, %1, %2 offset:0"
                                 :: "v"(dst), "v"(src), "s"(wbase) : "memory");
                }
            }
            asm volatile("s_wait_asynccnt 0x0" ::: "memory");
            __syncthreads();                      // slab visible to all waves

            AF a;
            {
                int row  = 16 * wm + ln15;
                int base = (ks * KB) >> 1;        // uint (K-pair) index
#pragma unroll
                for (int v = 0; v < 8; ++v) {
                    int koff = hi ? 8 : 0;
                    a.u[v] = sA[row][base + koff + v];
                }
            }
            // software-pipelined tile loop: load B[t+1] while WMMA t runs
            AF bc, bn;
            {
                const unsigned short* q = &sBs[(8 * wq) * (32 * LREC) + lane * LREC];
                uint4 x0 = *(const uint4*)q;
                uint4 x1 = *(const uint4*)(q + 8);
                bc.u[0]=x0.x; bc.u[1]=x0.y; bc.u[2]=x0.z; bc.u[3]=x0.w;
                bc.u[4]=x1.x; bc.u[5]=x1.y; bc.u[6]=x1.z; bc.u[7]=x1.w;
            }
#pragma unroll
            for (int t = 0; t < 8; ++t) {
                if (t < 7) {
                    const unsigned short* q = &sBs[(8 * wq + t + 1) * (32 * LREC) + lane * LREC];
                    uint4 x0 = *(const uint4*)q;
                    uint4 x1 = *(const uint4*)(q + 8);
                    bn.u[0]=x0.x; bn.u[1]=x0.y; bn.u[2]=x0.z; bn.u[3]=x0.w;
                    bn.u[4]=x1.x; bn.u[5]=x1.y; bn.u[6]=x1.z; bn.u[7]=x1.w;
                }
                acc[t] = __builtin_amdgcn_wmma_f32_16x16x32_bf16(
                    false, a.v, false, bc.v, (short)0, acc[t], false, false);
                if (t < 7) bc = bn;
            }
        }
    };

    // ================= GEMM1: h = x @ W1 =================
    run_gemm(e * NSLAB);

    // ---- + b_mlp ----
#pragma unroll
    for (int t = 0; t < 8; ++t) {
        int n = 128 * wq + 16 * t + ln15;
        float bias = bm[(size_t)e * DDIM + n];
#pragma unroll
        for (int r = 0; r < 8; ++r) acc[t][r] += bias;
    }

    // ================= LayerNorm directly from accumulators =================
    float psum[8], psq[8];
#pragma unroll
    for (int r = 0; r < 8; ++r) { psum[r] = 0.f; psq[r] = 0.f; }
#pragma unroll
    for (int t = 0; t < 8; ++t)
#pragma unroll
        for (int r = 0; r < 8; ++r) { float v = acc[t][r]; psum[r] += v; psq[r] += v * v; }
#pragma unroll
    for (int m = 1; m < 16; m <<= 1)
#pragma unroll
        for (int r = 0; r < 8; ++r) {
            psum[r] += __shfl_xor(psum[r], m, 32);
            psq[r]  += __shfl_xor(psq[r],  m, 32);
        }
    __syncthreads();               // A-panel reads done; safe to reuse sA below
    if (ln15 == 0) {
#pragma unroll
        for (int r = 0; r < 8; ++r) {
            int row = 16 * wm + r + 8 * hi;
            sPart[row][wq][0] = psum[r];
            sPart[row][wq][1] = psq[r];
        }
    }
    __syncthreads();
    if (tid < ROWS) {
        float s = 0.f, q = 0.f;
#pragma unroll
        for (int j = 0; j < 4; ++j) { s += sPart[tid][j][0]; q += sPart[tid][j][1]; }
        float mu  = s * (1.f / DDIM);
        float var = q * (1.f / DDIM) - mu * mu;
        sMu[tid]   = mu;
        sRstd[tid] = rsqrtf(var + 1e-5f);
    }
    __syncthreads();

    // ---- normalize + relu -> bf16, overwrite A panel ----
    {
        unsigned short* sAh = (unsigned short*)&sA[0][0];
        const int rstride = (DDIM / 2 + 2) * 2;
#pragma unroll
        for (int t = 0; t < 8; ++t) {
            int n = 128 * wq + 16 * t + ln15;
            float g = sG[n], bb = sBt[n];
#pragma unroll
            for (int r = 0; r < 8; ++r) {
                int row = 16 * wm + r + 8 * hi;
                float v = (acc[t][r] - sMu[row]) * sRstd[row] * g + bb;
                v = fmaxf(v, 0.f);
                sAh[row * rstride + n] = f2bf(v);
            }
        }
    }

    // ================= GEMM2: out = h @ W_out + b_out =================
#pragma unroll
    for (int t = 0; t < 8; ++t)
#pragma unroll
        for (int r = 0; r < 8; ++r) acc[t][r] = 0.f;

    run_gemm(EXPERTS * NSLAB + e * NSLAB);

    float* orow = out + ((size_t)e * NTOK + row0) * ODIM;
#pragma unroll
    for (int t = 0; t < 8; ++t) {
        int n = 128 * wq + 16 * t + ln15;
        float bias = bo[(size_t)e * ODIM + n];
#pragma unroll
        for (int r = 0; r < 8; ++r) {
            int row = 16 * wm + r + 8 * hi;
            orow[(size_t)row * ODIM + n] = acc[t][r] + bias;
        }
    }
}

extern "C" void kernel_launch(void* const* d_in, const int* in_sizes, int n_in,
                              void* d_out, int out_size, void* d_ws, size_t ws_size,
                              hipStream_t stream) {
    (void)in_sizes; (void)n_in; (void)out_size; (void)ws_size;   // needs >= 16 MiB ws
    const float* x  = (const float*)d_in[0];
    const float* Wm = (const float*)d_in[1];
    const float* bm = (const float*)d_in[2];
    const float* Wo = (const float*)d_in[3];
    const float* bo = (const float*)d_in[4];
    const float* g  = (const float*)d_in[5];
    const float* b  = (const float*)d_in[6];
    unsigned short* ws = (unsigned short*)d_ws;

    const size_t total = 2ull * EXPERTS * DDIM * DDIM;
    hipLaunchKernelGGL(swizzle_weights, dim3((unsigned)((total + 255) / 256)), dim3(256),
                       0, stream, Wm, Wo, ws);

    dim3 grid(NTOK / ROWS, EXPERTS);
    hipLaunchKernelGGL(moe_mlp_fused, grid, dim3(256), 0, stream,
                       x, (const unsigned short*)ws, bm, bo, g, b, (float*)d_out);
}
